// MultiHeadAttention_17669495456035
// MI455X (gfx1250) — matrix operand order repaired and checked by reading, not compiled, not run on hardware
//
#include <hip/hip_runtime.h>
#include <hip/hip_fp16.h>

typedef __attribute__((ext_vector_type(16))) _Float16 v16h;
typedef __attribute__((ext_vector_type(8)))  float    v8f;

#define B_SZ    2
#define S_LEN   2048
#define DMODEL  1024
#define NHEAD   16
#define DHEAD   64
#define HD      (NHEAD * DHEAD)   // 1024
#define M_TOT   (B_SZ * S_LEN)    // 4096
#define NEGBIG  (-1e12f)

// ---------------------------------------------------------------------------
// CDNA5 async global->LDS copy (16B per lane), ASYNCcnt-tracked.
// GV addressing: VDST = LDS byte address, VADDR = 64-bit global address.
// ---------------------------------------------------------------------------
__device__ __forceinline__ void async_copy_b128(void* lds_dst, const void* gsrc) {
    uint32_t l = (uint32_t)(uintptr_t)lds_dst;   // LDS aperture: low 32 bits
    asm volatile("global_load_async_to_lds_b128 %0, %1, off"
                 :: "v"(l), "v"(gsrc) : "memory");
}

__device__ __forceinline__ void wait_asynccnt0() {
#if __has_builtin(__builtin_amdgcn_s_wait_asynccnt)
    __builtin_amdgcn_s_wait_asynccnt(0);
#else
    asm volatile("s_wait_asynccnt 0x0" ::: "memory");
#endif
}

// ---------------------------------------------------------------------------
// f32 -> f16 conversion (x)
// ---------------------------------------------------------------------------
__global__ void f32_to_f16_kernel(const float* __restrict__ src,
                                  _Float16* __restrict__ dst, int n) {
    int i = blockIdx.x * blockDim.x + threadIdx.x;
    if (i < n) dst[i] = (_Float16)src[i];
}

// ---------------------------------------------------------------------------
// f32 -> f16 transposing conversion for weights: dst[n*K + k] = src[k*N + n]
// ---------------------------------------------------------------------------
__global__ void __launch_bounds__(256)
f32_to_f16_transpose_kernel(const float* __restrict__ src,
                            _Float16* __restrict__ dst, int K, int N) {
    __shared__ float tile[32][33];
    const int n0 = blockIdx.x * 32;
    const int k0 = blockIdx.y * 32;
    const int tx = threadIdx.x & 31;
    const int ty = threadIdx.x >> 5;   // 0..7
    #pragma unroll
    for (int i = 0; i < 32; i += 8)
        tile[ty + i][tx] = src[(size_t)(k0 + ty + i) * N + n0 + tx];
    __syncthreads();
    #pragma unroll
    for (int i = 0; i < 32; i += 8)
        dst[(size_t)(n0 + ty + i) * K + k0 + tx] = (_Float16)tile[tx][ty + i];
}

// ---------------------------------------------------------------------------
// Tiled WMMA GEMM: C[M,N] = A[M,K] (f16, row-major) * BT[N,K] (f16, N-major!)
//   mode 0: RoPE epilogue, store f16 [M,N]          (Q, K projections)
//   mode 1: store f16 transposed [B,H,DH,S]         (V projection)
//   mode 2: store f32 + bias [M,N]                  (final projection)
// Block = 128 threads (4 waves). Block tile 64x64, K-chunk 32.
// ---------------------------------------------------------------------------
__global__ void __launch_bounds__(128)
gemm_wmma_kernel(const _Float16* __restrict__ A, const _Float16* __restrict__ BT,
                 const float* __restrict__ bias, const float* __restrict__ pos,
                 _Float16* __restrict__ outH, float* __restrict__ outF,
                 int M, int N, int K, int mode)
{
    __shared__ _Float16 Ah[64 * 32];    // [row][k]
    __shared__ _Float16 BhT[64 * 32];   // [n][k]  (N-major)

    const int tid    = threadIdx.x;
    const int lane   = tid & 31;
    const int wave   = tid >> 5;
    const int laneHi = (lane >= 16) ? 1 : 0;
    const int lane16 = lane & 15;
    const int m0 = blockIdx.y * 64;
    const int n0 = blockIdx.x * 64;

    v8f acc[4];
    #pragma unroll
    for (int i = 0; i < 4; ++i) acc[i] = {};

    for (int kc = 0; kc < K; kc += 32) {
        __syncthreads();
        // Stage A tile (64 rows x 32 k) and BT tile (64 n x 32 k)
        #pragma unroll
        for (int i = 0; i < 2; ++i) {
            int flat8 = tid + i * 128;
            int r = flat8 >> 2;          // 0..63
            int c = (flat8 & 3) * 8;     // 0,8,16,24
            *(uint4*)&Ah[r * 32 + c] =
                *(const uint4*)&A[(size_t)(m0 + r) * K + kc + c];
            *(uint4*)&BhT[r * 32 + c] =
                *(const uint4*)&BT[(size_t)(n0 + r) * K + kc + c];
        }
        if (kc + 32 < K) {  // global_prefetch_b8 for next K tile
            __builtin_prefetch(&A[(size_t)(m0 + (tid >> 2)) * K + kc + 32], 0, 1);
            __builtin_prefetch(&BT[(size_t)(n0 + (tid >> 2)) * K + kc + 32], 0, 1);
        }
        __syncthreads();

        // A fragment (16x32 f16): 16 contiguous halves per lane
        v16h af;
        #pragma unroll
        for (int h = 0; h < 16; ++h) {
            int kk = h + (laneHi ? 16 : 0);
            af[h] = Ah[(wave * 16 + lane16) * 32 + kk];
        }
        #pragma unroll
        for (int ct = 0; ct < 4; ++ct) {
            // B fragment (32x16): 16 contiguous halves per lane in BhT
            v16h bf;
            #pragma unroll
            for (int h = 0; h < 16; ++h)
                bf[h] = BhT[(ct * 16 + lane16) * 32 + laneHi * 16 + h];
            acc[ct] = __builtin_amdgcn_wmma_f32_16x16x32_f16(
                false, af, false, bf, (short)0, acc[ct], false, false);
        }
    }

    // Epilogue. C layout: VGPR r -> row = r + 8*laneHi, col = ct*16 + lane%16
    #pragma unroll
    for (int ct = 0; ct < 4; ++ct) {
        #pragma unroll
        for (int r = 0; r < 8; ++r) {
            int row = m0 + wave * 16 + r + (laneHi ? 8 : 0);
            int col = n0 + ct * 16 + lane16;
            float v = acc[ct][r] + bias[col];
            if (mode == 0) {
                // RoPE: cos[d]=pos[s, d|1], sin[d]=pos[s, d&~1]
                // t2[d] = (d even) ? -t[d+1] : t[d-1]; partner lane = lane^1
                int s = row & (S_LEN - 1);
                int d = col & (DHEAD - 1);
                float cosv = pos[s * DHEAD + (d | 1)];
                float sinv = pos[s * DHEAD + (d & ~1)];
                float partner = __shfl_xor(v, 1);
                float rot = v * cosv + ((d & 1) ? partner : -partner) * sinv;
                outH[(size_t)row * N + col] = (_Float16)rot;
            } else if (mode == 1) {
                // V stored transposed: [B, H, DH, S]
                int b = row >> 11;            // row / S_LEN
                int s = row & (S_LEN - 1);
                int h = col >> 6;             // col / DHEAD
                int d = col & (DHEAD - 1);
                outH[(size_t)((b * NHEAD + h) * DHEAD + d) * S_LEN + s] =
                    (_Float16)v;
            } else {
                outF[(size_t)row * N + col] = v;
            }
        }
    }
}

// ---------------------------------------------------------------------------
// Flash attention: one block = (64 q rows) x (one head, one batch).
// 4 waves; each wave owns 16 q rows. Streams K/V in 32-key tiles staged with
// CDNA5 async global->LDS copies (ASYNCcnt).
// ---------------------------------------------------------------------------
__global__ void __launch_bounds__(128)
attn_wmma_kernel(const _Float16* __restrict__ Qh, const _Float16* __restrict__ Kh,
                 const _Float16* __restrict__ VhT, const int* __restrict__ vmask,
                 _Float16* __restrict__ Oh)
{
    __shared__ _Float16 Qt[64 * 64];    // [q][dh]
    __shared__ _Float16 Kt[32 * 64];    // [key][dh]
    __shared__ _Float16 VtT[64 * 32];   // [dh][key]  (transposed)
    __shared__ _Float16 Pb[4][16 * 32]; // per-wave P staging [q][key]

    const int tid    = threadIdx.x;
    const int lane   = tid & 31;
    const int wave   = tid >> 5;
    const int laneHi = (lane >= 16) ? 1 : 0;
    const int lane16 = lane & 15;
    const int qb = blockIdx.x;   // S/64 query blocks
    const int hh = blockIdx.y;   // head
    const int bb = blockIdx.z;   // batch
    const size_t strideS = HD;   // elements per (b,s) row

    // Load 64x64 Q tile
    #pragma unroll
    for (int i = 0; i < 4; ++i) {
        int flat8 = tid + i * 128;
        int r = flat8 >> 3;
        int c = (flat8 & 7) * 8;
        *(uint4*)&Qt[r * 64 + c] =
            *(const uint4*)&Qh[(size_t)(bb * S_LEN + qb * 64 + r) * strideS +
                               hh * DHEAD + c];
    }
    __syncthreads();

    // Q A-fragments in registers (DH=64 -> two K=32 chunks)
    v16h aq[2];
    #pragma unroll
    for (int kc = 0; kc < 2; ++kc)
        #pragma unroll
        for (int h = 0; h < 16; ++h) {
            int kk = kc * 32 + h + (laneHi ? 16 : 0);
            aq[kc][h] = Qt[(wave * 16 + lane16) * 64 + kk];
        }

    float mrun[8], lrun[8];
    v8f acc[4];
    #pragma unroll
    for (int r = 0; r < 8; ++r) { mrun[r] = -3.0e38f; lrun[r] = 0.0f; }
    #pragma unroll
    for (int i = 0; i < 4; ++i) acc[i] = {};

    for (int kb = 0; kb < S_LEN / 32; ++kb) {
        __syncthreads();
        // Async-stage K tile [32 key][64 dh] from [B,S,H,DH]
        #pragma unroll
        for (int i = 0; i < 2; ++i) {
            int flat8 = tid + i * 128;
            int r = flat8 >> 3;          // key 0..31
            int c = (flat8 & 7) * 8;     // dh
            async_copy_b128(&Kt[r * 64 + c],
                            &Kh[(size_t)(bb * S_LEN + kb * 32 + r) * strideS +
                                hh * DHEAD + c]);
        }
        // Async-stage V^T tile [64 dh][32 key] from [B,H,DH,S]
        #pragma unroll
        for (int i = 0; i < 2; ++i) {
            int flat8 = tid + i * 128;
            int d = flat8 >> 2;          // dh 0..63
            int c = (flat8 & 3) * 8;     // key 0,8,16,24
            async_copy_b128(&VtT[d * 32 + c],
                            &VhT[(size_t)((bb * NHEAD + hh) * DHEAD + d) * S_LEN +
                                 kb * 32 + c]);
        }
        wait_asynccnt0();
        __syncthreads();

        // Scores: s0 = keys[0..15], s1 = keys[16..31] of this block
        v8f s0 = {}, s1 = {};
        #pragma unroll
        for (int kc = 0; kc < 2; ++kc) {
            v16h bk0, bk1;
            #pragma unroll
            for (int h = 0; h < 16; ++h) {
                int dd = kc * 32 + laneHi * 16 + h;          // contiguous in h
                bk0[h] = Kt[lane16 * 64 + dd];               // key = lane16
                bk1[h] = Kt[(16 + lane16) * 64 + dd];        // key = 16+lane16
            }
            s0 = __builtin_amdgcn_wmma_f32_16x16x32_f16(
                false, aq[kc], false, bk0, (short)0, s0, false, false);
            s1 = __builtin_amdgcn_wmma_f32_16x16x32_f16(
                false, aq[kc], false, bk1, (short)0, s1, false, false);
        }

        float mf0 = (float)vmask[bb * S_LEN + kb * 32 + lane16];
        float mf1 = (float)vmask[bb * S_LEN + kb * 32 + 16 + lane16];

        // Online softmax, rows striped per the C layout (row = r + 8*laneHi)
        #pragma unroll
        for (int r = 0; r < 8; ++r) {
            float a0 = s0[r] * 0.125f;  a0 = a0 * mf0 + NEGBIG * (1.0f - mf0);
            float a1 = s1[r] * 0.125f;  a1 = a1 * mf1 + NEGBIG * (1.0f - mf1);
            float bm = fmaxf(a0, a1);
            #pragma unroll
            for (int off = 1; off < 16; off <<= 1)
                bm = fmaxf(bm, __shfl_xor(bm, off));
            float mnew = fmaxf(mrun[r], bm);
            float corr = __expf(mrun[r] - mnew);
            float p0 = __expf(a0 - mnew);
            float p1 = __expf(a1 - mnew);
            float ps = p0 + p1;
            #pragma unroll
            for (int off = 1; off < 16; off <<= 1)
                ps += __shfl_xor(ps, off);
            lrun[r] = lrun[r] * corr + ps;
            mrun[r] = mnew;
            #pragma unroll
            for (int ct = 0; ct < 4; ++ct) acc[ct][r] *= corr;
            int prow = r + (laneHi ? 8 : 0);
            Pb[wave][prow * 32 + lane16]      = (_Float16)p0;
            Pb[wave][prow * 32 + 16 + lane16] = (_Float16)p1;
        }
        __builtin_amdgcn_wave_barrier();  // same-wave LDS RAW: keep order

        // Re-read P as an A fragment (16x32): 16 contiguous halves
        v16h pf;
        #pragma unroll
        for (int h = 0; h < 16; ++h) {
            int kk = h + (laneHi ? 16 : 0);
            pf[h] = Pb[wave][lane16 * 32 + kk];
        }
        // acc += P (16x32) * V (32x64); V fragment from VtT is contiguous
        #pragma unroll
        for (int ct = 0; ct < 4; ++ct) {
            v16h bv;
            #pragma unroll
            for (int h = 0; h < 16; ++h)
                bv[h] = VtT[(ct * 16 + lane16) * 32 + laneHi * 16 + h];
            acc[ct] = __builtin_amdgcn_wmma_f32_16x16x32_f16(
                false, pf, false, bv, (short)0, acc[ct], false, false);
        }
    }

    // Normalize and store f16 intermediate [B,S,H*DH]
    #pragma unroll
    for (int r = 0; r < 8; ++r) {
        float inv = 1.0f / lrun[r];
        int q = qb * 64 + wave * 16 + r + (laneHi ? 8 : 0);
        #pragma unroll
        for (int ct = 0; ct < 4; ++ct)
            Oh[(size_t)(bb * S_LEN + q) * strideS + hh * DHEAD + ct * 16 + lane16] =
                (_Float16)(acc[ct][r] * inv);
    }
}

// ---------------------------------------------------------------------------
// Launch
// ---------------------------------------------------------------------------
extern "C" void kernel_launch(void* const* d_in, const int* in_sizes, int n_in,
                              void* d_out, int out_size, void* d_ws, size_t ws_size,
                              hipStream_t stream) {
    const float* x     = (const float*)d_in[0];
    const int*   vmask = (const int*)d_in[1];
    const float* pos   = (const float*)d_in[2];
    const float* Wq    = (const float*)d_in[3];
    const float* bq    = (const float*)d_in[4];
    const float* Wk    = (const float*)d_in[5];
    const float* bk    = (const float*)d_in[6];
    const float* Wv    = (const float*)d_in[7];
    const float* bv    = (const float*)d_in[8];
    const float* Wo    = (const float*)d_in[9];
    const float* bo    = (const float*)d_in[10];
    float* out = (float*)d_out;

    // Workspace (f16): xh 8MB | WqT/WkT/WvT/WoT 2MB each | Q/K/V/O 8MB each
    char* ws = (char*)d_ws;
    _Float16* xh   = (_Float16*)(ws);
    _Float16* WqT  = (_Float16*)(ws + ( 8u << 20));
    _Float16* WkT  = (_Float16*)(ws + (10u << 20));
    _Float16* WvT  = (_Float16*)(ws + (12u << 20));
    _Float16* WoT  = (_Float16*)(ws + (14u << 20));
    _Float16* Qhh  = (_Float16*)(ws + (16u << 20));
    _Float16* Khh  = (_Float16*)(ws + (24u << 20));
    _Float16* VhhT = (_Float16*)(ws + (32u << 20));
    _Float16* Ohh  = (_Float16*)(ws + (40u << 20));

    const int nx = M_TOT * DMODEL;     // 4M
    f32_to_f16_kernel<<<(nx + 255) / 256, 256, 0, stream>>>(x, xh, nx);

    dim3 gT(HD / 32, DMODEL / 32);     // (32, 32)
    f32_to_f16_transpose_kernel<<<gT, 256, 0, stream>>>(Wq, WqT, DMODEL, HD);
    f32_to_f16_transpose_kernel<<<gT, 256, 0, stream>>>(Wk, WkT, DMODEL, HD);
    f32_to_f16_transpose_kernel<<<gT, 256, 0, stream>>>(Wv, WvT, DMODEL, HD);
    f32_to_f16_transpose_kernel<<<gT, 256, 0, stream>>>(Wo, WoT, HD, DMODEL);

    dim3 gGemm(HD / 64, M_TOT / 64);   // (16, 64)
    gemm_wmma_kernel<<<gGemm, 128, 0, stream>>>(xh, WqT, bq, pos, Qhh, nullptr,
                                                M_TOT, HD, DMODEL, 0);
    gemm_wmma_kernel<<<gGemm, 128, 0, stream>>>(xh, WkT, bk, pos, Khh, nullptr,
                                                M_TOT, HD, DMODEL, 0);
    gemm_wmma_kernel<<<gGemm, 128, 0, stream>>>(xh, WvT, bv, pos, VhhT, nullptr,
                                                M_TOT, HD, DMODEL, 1);

    dim3 gAttn(S_LEN / 64, NHEAD, B_SZ);  // (32, 16, 2)
    attn_wmma_kernel<<<gAttn, 128, 0, stream>>>(Qhh, Khh, VhhT, vmask, Ohh);

    gemm_wmma_kernel<<<gGemm, 128, 0, stream>>>(Ohh, WoT, bo, pos, nullptr, out,
                                                M_TOT, DMODEL, HD, 2);
}
